// PFGRUCell_20418274525681
// MI455X (gfx1250) — matrix operand order repaired and checked by reading, not compiled, not run on hardware
//
#include <hip/hip_runtime.h>
#include <stdint.h>

#define K_PART 16384
#define IN_DIM 512
#define H_DIM  1024
#define CAT    1536      // H + IN
#define H2     2048
#define ALPHAC 0.5f
#define ASTRIDE 1544     // CAT + 8 bf16 pad -> LDS row stride 3088B, kills bank conflicts

typedef __attribute__((ext_vector_type(16))) __bf16   v16bf;
typedef __attribute__((ext_vector_type(8)))  float    v8f;
typedef __attribute__((ext_vector_type(4)))  uint32_t u32x4;

struct Frag32B { u32x4 a, b; };   // 32-byte POD, bit-cast to v16bf
struct BP { v16bf z, r; };        // one (w_z, w_r) fragment pair
struct NP { v16bf m, v; };        // one (mu, var) fragment pair

// f32 -> bf16 round-to-nearest-even
__device__ __forceinline__ uint16_t f2bf(float x) {
  uint32_t b = __float_as_uint(x);
  uint32_t r = b + 0x7FFFu + ((b >> 16) & 1u);
  return (uint16_t)(r >> 16);
}

__device__ __forceinline__ v16bf load_frag_g(const uint16_t* __restrict__ p) {
  Frag32B t;
  t.a = *(const u32x4*)p;
  t.b = *(const u32x4*)(p + 8);
  return __builtin_bit_cast(v16bf, t);
}

__device__ __forceinline__ v16bf load_frag_lds(const uint16_t* p0, const uint16_t* p1) {
  Frag32B t;
  t.a = *(const u32x4*)p0;
  t.b = *(const u32x4*)p1;
  return __builtin_bit_cast(v16bf, t);
}

__device__ __forceinline__ v8f zero8() {
  v8f r;
  #pragma unroll
  for (int i = 0; i < 8; ++i) r[i] = 0.f;
  return r;
}

// ---------------------------------------------------------------------------
// Kernel 0: convert f32 weights to bf16 pre-swizzled into WMMA B-fragment
// layout. Tile = 32(K) x 16(N) = 512 bf16. Element (k,n):
//   lane = (n&15) | (k&16), slot = k&15  -> lane's 16 bf16 are contiguous.
// GEMM kernel then loads a whole B fragment as lane-contiguous 32B.
// ---------------------------------------------------------------------------
__global__ __launch_bounds__(256) void swizzle_kernel(
    const float* __restrict__ W, uint16_t* __restrict__ out,
    int total, int lgN) {
  int e = blockIdx.x * 256 + threadIdx.x;
  if (e >= total) return;
  int k = e >> lgN;
  int n = e & ((1 << lgN) - 1);
  int ntiles = 1 << (lgN - 4);
  int kt = k >> 5, kk = k & 31;
  int nt = n >> 4, nn = n & 15;
  int lane = nn | (kk & 16);
  int slot = kk & 15;
  out[(((size_t)(kt * ntiles + nt)) << 9) | (lane << 4) | slot] = f2bf(W[e]);
}

// ---------------------------------------------------------------------------
// Kernel 1: fused GRU GEMMs.  512 blocks x 512 threads (16 waves).
// Block owns 32 particle rows. Wave w: rowHalf = w>>3, colGroup = w&7,
// column tiles ct = colGroup + 8*j, j=0..7.  B fragments are prefetched
// one j-step ahead through a depth-2 ring (32 VGPRs) so each WMMA pair
// waits on loads issued a full step earlier (s_wait_loadcnt > 0), while
// the 128 accumulator VGPRs stay within the 4-waves/SIMD budget.
//   phase 1: A=[h0|obs] bf16 in LDS; z,r = sigmoid(A@w_z/r) via WMMA
//   phase 2: overwrite LDS cols 0..1023 with bf16(r*h0)
//   phase 3: mu/var pair GEMMs + fused tanh/softplus/GRU blend -> h1
// ---------------------------------------------------------------------------
__global__ __launch_bounds__(512) void gru_gemm_kernel(
    const float* __restrict__ input_, const float* __restrict__ h0,
    const float* __restrict__ eps,
    const float* __restrict__ b_z, const float* __restrict__ b_r,
    const float* __restrict__ b_n,
    const uint16_t* __restrict__ wz, const uint16_t* __restrict__ wr,
    const uint16_t* __restrict__ wn,
    float* __restrict__ h1_out) {
  extern __shared__ uint16_t ldsA[];          // 32 x ASTRIDE bf16
  const int tid = threadIdx.x;
  const int rowBase = blockIdx.x * 32;

  // Build A = [h0 | obs] (bf16), coalesced over columns.
  for (int e = tid; e < 32 * CAT; e += 512) {
    int r = e / CAT, c = e - r * CAT;
    float v = (c < H_DIM) ? h0[(size_t)(rowBase + r) * H_DIM + c]
                          : input_[c - H_DIM];
    ldsA[r * ASTRIDE + c] = f2bf(v);
  }
  __syncthreads();

  const int wave = tid >> 5, lane = tid & 31;
  const int rowHalf = wave >> 3;              // 0 or 1
  const int colGrp  = wave & 7;
  const int am    = (rowHalf << 4) + (lane & 15);   // A-fragment row
  const int aksel = (lane >> 4) << 4;               // 0 or 16 (K select)
  const int bOff  = lane << 4;                      // lane*16 elems into B tile
  const int cn    = lane & 15;                      // C-frag column-in-tile
  const int rv    = (lane >> 4) << 3;               // C-frag +8 row for lanes 16..31

  auto loadA = [&](int kt) -> v16bf {
    const uint16_t* ar = &ldsA[am * ASTRIDE + (kt << 5) + aksel];
    return load_frag_lds(ar, ar + 8);
  };
  auto loadBP = [&](BP& b, int kt, int j) {
    const size_t toff = ((size_t)((kt << 6) + colGrp + (j << 3)) << 9) + bOff;
    b.z = load_frag_g(wz + toff);
    b.r = load_frag_g(wr + toff);
  };

  v8f zac[8], rac[8];
  #pragma unroll
  for (int j = 0; j < 8; ++j) { zac[j] = zero8(); rac[j] = zero8(); }

  // ---- z / r GEMMs: j-step ring prefetch (8 even -> ring phase stable) ----
  {
    BP ring[2];
    loadBP(ring[0], 0, 0);
    #pragma unroll 1
    for (int kt = 0; kt < 48; ++kt) {
      v16bf a = loadA(kt);
      #pragma unroll
      for (int j = 0; j < 8; ++j) {
        const int jn = j + 1;
        if (jn < 8) {
          loadBP(ring[jn & 1], kt, jn);
        } else if (kt + 1 < 48) {
          loadBP(ring[jn & 1], kt + 1, 0);
        }
        zac[j] = __builtin_amdgcn_wmma_f32_16x16x32_bf16(
            false, a, false, ring[j & 1].z, (short)0, zac[j], false, false);
        rac[j] = __builtin_amdgcn_wmma_f32_16x16x32_bf16(
            false, a, false, ring[j & 1].r, (short)0, rac[j], false, false);
      }
    }
  }

  // sigmoid + bias in registers
  #pragma unroll
  for (int j = 0; j < 8; ++j) {
    const int ct = colGrp + (j << 3);
    const float bz = b_z[(ct << 4) + cn];
    const float br = b_r[(ct << 4) + cn];
    #pragma unroll
    for (int v = 0; v < 8; ++v) {
      zac[j][v] = 1.f / (1.f + __expf(-(zac[j][v] + bz)));
      rac[j][v] = 1.f / (1.f + __expf(-(rac[j][v] + br)));
    }
  }
  __syncthreads();

  // ---- overwrite A cols [0,1024) with bf16(r * h0); obs half is reused ----
  #pragma unroll
  for (int j = 0; j < 8; ++j) {
    const int ct = colGrp + (j << 3);
    const int col = (ct << 4) + cn;
    #pragma unroll
    for (int v = 0; v < 8; ++v) {
      const int rl = (rowHalf << 4) + rv + v;
      float h0v = h0[(size_t)(rowBase + rl) * H_DIM + col];
      ldsA[rl * ASTRIDE + col] = f2bf(rac[j][v] * h0v);
    }
  }
  __syncthreads();

  // ---- n GEMM: mu (col-tile ct) + var (col-tile ct+64) pairs ----
  #pragma unroll 1
  for (int p = 0; p < 8; ++p) {
    const int ct = colGrp + (p << 3);
    v8f mac = zero8(), vac = zero8();
    auto loadN = [&](NP& nb, int kt) {
      const size_t base = ((size_t)((kt << 7) + ct) << 9) + bOff;
      nb.m = load_frag_g(wn + base);
      nb.v = load_frag_g(wn + base + ((size_t)64 << 9));
    };
    NP ring[2];
    loadN(ring[0], 0);
    #pragma unroll 1
    for (int kt = 0; kt < 48; ++kt) {
      v16bf a = loadA(kt);
      if (kt + 1 < 48) loadN(ring[(kt + 1) & 1], kt + 1);
      const NP& cur = ring[kt & 1];
      mac = __builtin_amdgcn_wmma_f32_16x16x32_bf16(
          false, a, false, cur.m, (short)0, mac, false, false);
      vac = __builtin_amdgcn_wmma_f32_16x16x32_bf16(
          false, a, false, cur.v, (short)0, vac, false, false);
    }

    const int col = (ct << 4) + cn;
    const float bm_ = b_n[col], bv_ = b_n[H_DIM + col];
    #pragma unroll
    for (int v = 0; v < 8; ++v) {
      const int rl = (rowHalf << 4) + rv + v;
      const size_t g = (size_t)(rowBase + rl) * H_DIM + col;
      float e   = eps[g];
      float h0v = h0[g];
      float mu  = mac[v] + bm_;
      float va  = vac[v] + bv_;
      float sp  = (va > 15.f) ? va : log1pf(__expf(va));
      float n   = tanhf(mu + e * sp);
      float zv  = zac[p][v];
      h1_out[g] = (1.f - zv) * n + zv * h0v;
    }
  }
}

// ---------------------------------------------------------------------------
// Kernel 2: logits[k] = [h1,obs] . w_obs + b_obs + p0[k]   (one wave per row)
// ---------------------------------------------------------------------------
__global__ __launch_bounds__(256) void logits_kernel(
    const float* __restrict__ h1, const float* __restrict__ input_,
    const float* __restrict__ w_obs, const float* __restrict__ b_obs,
    const float* __restrict__ p0, float* __restrict__ logits) {
  int wave = threadIdx.x >> 5, lane = threadIdx.x & 31;
  int row = blockIdx.x * 8 + wave;
  float s = 0.f;
  for (int c = lane; c < CAT; c += 32) {
    float x = (c < H_DIM) ? h1[(size_t)row * H_DIM + c] : input_[c - H_DIM];
    s += x * w_obs[c];
  }
  #pragma unroll
  for (int off = 16; off; off >>= 1) s += __shfl_xor(s, off, 32);
  if (lane == 0) logits[row] = s + b_obs[0] + p0[row];
}

// ---------------------------------------------------------------------------
// Kernel 3: single block: log_softmax over K, resampling-prob CDF (inclusive
// scan), zero mean_hid accumulator.
// ---------------------------------------------------------------------------
__global__ __launch_bounds__(1024) void softmax_cdf_kernel(
    const float* __restrict__ logits, float* __restrict__ p1t,
    float* __restrict__ cdf, float* __restrict__ mean_hid,
    float* __restrict__ totalOut) {
  __shared__ float red[1024];
  const int t = threadIdx.x;
  float m = -3.4e38f;
  for (int i = t; i < K_PART; i += 1024) m = fmaxf(m, logits[i]);
  red[t] = m; __syncthreads();
  for (int s = 512; s; s >>= 1) { if (t < s) red[t] = fmaxf(red[t], red[t + s]); __syncthreads(); }
  m = red[0]; __syncthreads();
  float sum = 0.f;
  for (int i = t; i < K_PART; i += 1024) sum += __expf(logits[i] - m);
  red[t] = sum; __syncthreads();
  for (int s = 512; s; s >>= 1) { if (t < s) red[t] += red[t + s]; __syncthreads(); }
  const float lse = m + __logf(red[0]);
  __syncthreads();

  // 16 contiguous elements per thread: local inclusive prefix of resamp prob
  float loc[16];
  float run = 0.f;
  #pragma unroll
  for (int j = 0; j < 16; ++j) {
    int i = t * 16 + j;
    float p = logits[i] - lse;
    p1t[i] = p;
    run += ALPHAC * __expf(p) + (1.f - ALPHAC) / K_PART;
    loc[j] = run;
  }
  red[t] = run; __syncthreads();
  // Hillis-Steele inclusive scan over thread totals
  for (int s = 1; s < 1024; s <<= 1) {
    float v = red[t];
    float add = (t >= s) ? red[t - s] : 0.f;
    __syncthreads();
    red[t] = v + add;
    __syncthreads();
  }
  float excl = (t > 0) ? red[t - 1] : 0.f;
  #pragma unroll
  for (int j = 0; j < 16; ++j) cdf[t * 16 + j] = loc[j] + excl;
  if (t == 0) totalOut[0] = red[1023];
  mean_hid[t] = 0.f;
}

// ---------------------------------------------------------------------------
// Kernel 4: systematic inverse-CDF resampling + per-sample log weight
// ---------------------------------------------------------------------------
__global__ __launch_bounds__(256) void resample_kernel(
    const float* __restrict__ cdf, const float* __restrict__ p1t,
    const float* __restrict__ totalIn, int* __restrict__ idx,
    float* __restrict__ pnl) {
  int i = blockIdx.x * 256 + threadIdx.x;
  float u = ((float)i + 0.5f) * (totalIn[0] / (float)K_PART);
  int lo = 0, hi = K_PART - 1;
  while (lo < hi) { int mid = (lo + hi) >> 1; if (cdf[mid] < u) lo = mid + 1; else hi = mid; }
  idx[i] = lo;
  float pe = __expf(p1t[lo]);
  float pn = pe / (ALPHAC * pe + (1.f - ALPHAC) / (float)K_PART);
  pnl[i] = __logf(pn);
}

// Kernel 5: single-block logsumexp over pnl -> lse2
__global__ __launch_bounds__(1024) void lse_kernel(
    const float* __restrict__ x, float* __restrict__ out) {
  __shared__ float red[1024];
  const int t = threadIdx.x;
  float m = -3.4e38f;
  for (int i = t; i < K_PART; i += 1024) m = fmaxf(m, x[i]);
  red[t] = m; __syncthreads();
  for (int s = 512; s; s >>= 1) { if (t < s) red[t] = fmaxf(red[t], red[t + s]); __syncthreads(); }
  m = red[0]; __syncthreads();
  float sum = 0.f;
  for (int i = t; i < K_PART; i += 1024) sum += __expf(x[i] - m);
  red[t] = sum; __syncthreads();
  for (int s = 512; s; s >>= 1) { if (t < s) red[t] += red[t + s]; __syncthreads(); }
  if (t == 0) out[0] = m + __logf(red[0]);
}

// ---------------------------------------------------------------------------
// Kernel 6: gather h1[idx] -> out, p1 out, weighted-mean partial reduction
// ---------------------------------------------------------------------------
__global__ __launch_bounds__(256) void gather_kernel(
    const float* __restrict__ h1t, const int* __restrict__ idx,
    const float* __restrict__ pnl, const float* __restrict__ lse2In,
    float* __restrict__ h1_out, float* __restrict__ p1_out,
    float* __restrict__ mean_hid) {
  const int t = threadIdx.x;
  const int r0 = blockIdx.x * 64;
  const float lse2 = lse2In[0];
  if (t < 64) p1_out[r0 + t] = pnl[r0 + t] - lse2;
  float a0 = 0.f, a1 = 0.f, a2 = 0.f, a3 = 0.f;
  for (int r = r0; r < r0 + 64; ++r) {
    int j = idx[r];
    float w = __expf(pnl[r] - lse2);
    const float* src = h1t + (size_t)j * H_DIM;
    float* dst = h1_out + (size_t)r * H_DIM;
    float v0 = src[t      ]; dst[t      ] = v0; a0 += w * v0;
    float v1 = src[t + 256]; dst[t + 256] = v1; a1 += w * v1;
    float v2 = src[t + 512]; dst[t + 512] = v2; a2 += w * v2;
    float v3 = src[t + 768]; dst[t + 768] = v3; a3 += w * v3;
  }
  atomicAdd(&mean_hid[t      ], a0);
  atomicAdd(&mean_hid[t + 256], a1);
  atomicAdd(&mean_hid[t + 512], a2);
  atomicAdd(&mean_hid[t + 768], a3);
}

// Kernel 7: tiny MLP 1024 -> 24 (relu) -> 2
__global__ __launch_bounds__(64) void mlp_kernel(
    const float* __restrict__ mean_hid,
    const float* __restrict__ w_h1, const float* __restrict__ b_h1,
    const float* __restrict__ w_h2, const float* __restrict__ b_h2,
    float* __restrict__ loc) {
  __shared__ float hid[24];
  const int t = threadIdx.x;
  if (t < 24) {
    float s = 0.f;
    for (int c = 0; c < H_DIM; ++c) s += mean_hid[c] * w_h1[c * 24 + t];
    hid[t] = fmaxf(s + b_h1[t], 0.f);
  }
  __syncthreads();
  if (t < 2) {
    float s = 0.f;
    for (int j = 0; j < 24; ++j) s += hid[j] * w_h2[j * 2 + t];
    loc[t] = s + b_h2[t];
  }
}

// ---------------------------------------------------------------------------
extern "C" void kernel_launch(void* const* d_in, const int* in_sizes, int n_in,
                              void* d_out, int out_size, void* d_ws, size_t ws_size,
                              hipStream_t stream) {
  const float* input_ = (const float*)d_in[0];
  const float* h0     = (const float*)d_in[1];
  const float* p0     = (const float*)d_in[2];
  const float* eps    = (const float*)d_in[3];
  const float* w_z    = (const float*)d_in[4];
  const float* b_z    = (const float*)d_in[5];
  const float* w_r    = (const float*)d_in[6];
  const float* b_r    = (const float*)d_in[7];
  const float* w_n    = (const float*)d_in[8];
  const float* b_n    = (const float*)d_in[9];
  const float* w_obs  = (const float*)d_in[10];
  const float* b_obs  = (const float*)d_in[11];
  const float* w_h1   = (const float*)d_in[12];
  const float* b_h1   = (const float*)d_in[13];
  const float* w_h2   = (const float*)d_in[14];
  const float* b_h2   = (const float*)d_in[15];

  char* ws = (char*)d_ws;
  size_t off = 0;
  uint16_t* wsz = (uint16_t*)(ws + off); off += (size_t)CAT * H_DIM * 2;
  uint16_t* wsr = (uint16_t*)(ws + off); off += (size_t)CAT * H_DIM * 2;
  uint16_t* wsn = (uint16_t*)(ws + off); off += (size_t)CAT * H2 * 2;
  float* h1t      = (float*)(ws + off);  off += (size_t)K_PART * H_DIM * 4;
  float* logits   = (float*)(ws + off);  off += (size_t)K_PART * 4;
  float* p1t      = (float*)(ws + off);  off += (size_t)K_PART * 4;
  float* cdf      = (float*)(ws + off);  off += (size_t)K_PART * 4;
  float* pnl      = (float*)(ws + off);  off += (size_t)K_PART * 4;
  int*   idx      = (int*)  (ws + off);  off += (size_t)K_PART * 4;
  float* mean_hid = (float*)(ws + off);  off += (size_t)H_DIM * 4;
  float* total    = (float*)(ws + off);  off += 256;
  float* lse2     = (float*)(ws + off);  off += 256;

  float* out_loc = (float*)d_out;
  float* out_h1  = out_loc + 2;
  float* out_p1  = out_h1 + (size_t)K_PART * H_DIM;

  const int nz = CAT * H_DIM, nn = CAT * H2;
  swizzle_kernel<<<(nz + 255) / 256, 256, 0, stream>>>(w_z, wsz, nz, 10);
  swizzle_kernel<<<(nz + 255) / 256, 256, 0, stream>>>(w_r, wsr, nz, 10);
  swizzle_kernel<<<(nn + 255) / 256, 256, 0, stream>>>(w_n, wsn, nn, 11);

  gru_gemm_kernel<<<K_PART / 32, 512, 32 * ASTRIDE * 2, stream>>>(
      input_, h0, eps, b_z, b_r, b_n, wsz, wsr, wsn, h1t);

  logits_kernel<<<K_PART / 8, 256, 0, stream>>>(h1t, input_, w_obs, b_obs, p0, logits);
  softmax_cdf_kernel<<<1, 1024, 0, stream>>>(logits, p1t, cdf, mean_hid, total);
  resample_kernel<<<K_PART / 256, 256, 0, stream>>>(cdf, p1t, total, idx, pnl);
  lse_kernel<<<1, 1024, 0, stream>>>(pnl, lse2);
  gather_kernel<<<K_PART / 64, 256, 0, stream>>>(h1t, idx, pnl, lse2, out_h1, out_p1, mean_hid);
  mlp_kernel<<<1, 64, 0, stream>>>(mean_hid, w_h1, b_h1, w_h2, b_h2, out_loc);
}
